// CombinedLoss_16037407883832
// MI455X (gfx1250) — compile-verified, matrix operand order checked by reading, not run on hardware
//
#include <hip/hip_runtime.h>

// Shapes are fixed by the reference: N=8, C=21, H=W=512  (HW = 2^18).
#define NCLS   21
#define LOG2HW 18

typedef __attribute__((ext_vector_type(16))) _Float16 v16h;
typedef __attribute__((ext_vector_type(8)))  float    v8f;

__device__ __forceinline__ float wave_red(float v) {
#pragma unroll
  for (int o = 16; o > 0; o >>= 1) v += __shfl_down(v, o, 32);
  return v;
}

// ws layout (65 floats): [0]=focal_sum [1]=ce_sum [2..22]=sum_probs[c]
//                        [23..43]=intersect[c]    [44..64]=counts[c]
__global__ void zero_ws(float* ws) {
  int i = threadIdx.x;
  if (i < 65) ws[i] = 0.0f;
}

__global__ __launch_bounds__(256) void loss_main(
    const float* __restrict__ logits, const float* __restrict__ cw,
    const int* __restrict__ tgt, float* __restrict__ ws, int nhw) {
  __shared__ float partLds[8][24];   // per-wave partials: focal, ce, sp[21]
  __shared__ float interLds[NCLS];
  __shared__ float cntLds[NCLS];
  __shared__ float cwLds[NCLS];

  const int tid = threadIdx.x;
  if (tid < NCLS) {
    interLds[tid] = 0.0f;
    cntLds[tid]   = 0.0f;
    cwLds[tid]    = cw[tid];
  }
  __syncthreads();

  float facc = 0.0f, ceacc = 0.0f;
  float sp[NCLS];
#pragma unroll
  for (int c = 0; c < NCLS; ++c) sp[c] = 0.0f;

  const int stride = gridDim.x * blockDim.x;
  for (int p = blockIdx.x * blockDim.x + tid; p < nhw; p += stride) {
    const int n  = p >> LOG2HW;
    const int hw = p & ((1 << LOG2HW) - 1);
    const float* base = logits + (((size_t)n * NCLS) << LOG2HW) + (size_t)hw;

    float l[NCLS];
#pragma unroll
    for (int c = 0; c < NCLS; ++c)
      l[c] = __builtin_nontemporal_load(base + ((size_t)c << LOG2HW));

    const int t = __builtin_nontemporal_load(tgt + p);

    float m = l[0];
#pragma unroll
    for (int c = 1; c < NCLS; ++c) m = fmaxf(m, l[c]);

    float lt = l[0];
#pragma unroll
    for (int c = 0; c < NCLS; ++c) lt = (c == t) ? l[c] : lt;

    float se = 0.0f;
#pragma unroll
    for (int c = 0; c < NCLS; ++c) { l[c] = __expf(l[c] - m); se += l[c]; }

    float et = l[0];
#pragma unroll
    for (int c = 0; c < NCLS; ++c) et = (c == t) ? l[c] : et;

    const float inv = 1.0f / se;
#pragma unroll
    for (int c = 0; c < NCLS; ++c) sp[c] += l[c] * inv;

    const float lpt = lt - m - __logf(se);
    const float ce  = -cwLds[t] * lpt;          // weighted CE, >= 0
    const float pt  = __expf(-ce);
    const float om  = 1.0f - pt;
    facc  += om * om * ce;                      // GAMMA = 2
    ceacc += ce;

    atomicAdd(&interLds[t], et * inv);          // ds_add_f32
    atomicAdd(&cntLds[t], 1.0f);
  }

  // ---- wave-level reduction of the 23 float accumulators ----
  const int lane = tid & 31, wv = tid >> 5;
  facc  = wave_red(facc);
  ceacc = wave_red(ceacc);
  float spr[NCLS];
#pragma unroll
  for (int c = 0; c < NCLS; ++c) spr[c] = wave_red(sp[c]);

  if (lane == 0) {
    partLds[wv][0] = facc;
    partLds[wv][1] = ceacc;
#pragma unroll
    for (int c = 0; c < NCLS; ++c) partLds[wv][2 + c] = spr[c];
  }
  __syncthreads();

  // ---- cross-wave reduction of the 8x23 table via WMMA ----
  // D = ones(16x32,f16) x B(32x16,f16): D[0][n] = sum_k B[k][n].
  // B (32x16, 16-bit), contiguous k order used here: lane<16 holds K=0..15 of column N=lane in its
  // 16 halves; lanes 16..31 hold K=16..31 (all zero here). Rows 0..7 carry
  // the 8 per-wave partials for column (tile*16 + lane).
  if (wv == 0) {
    v16h A;
#pragma unroll
    for (int k = 0; k < 16; ++k) A[k] = (_Float16)1.0f;
#pragma unroll
    for (int tile = 0; tile < 2; ++tile) {
      const int  col   = tile * 16 + (lane & 15);
      const bool valid = (lane < 16) && (col < 23);
      const int  ccol  = (col < 23) ? col : 0;
      v16h B;
#pragma unroll
      for (int k = 0; k < 16; ++k) {
        const float v = (k < 8) ? partLds[k][ccol] : 0.0f;
        B[k] = valid ? (_Float16)v : (_Float16)0.0f;
      }
      v8f Cz = {};
      v8f D = __builtin_amdgcn_wmma_f32_16x16x32_f16(
          /*neg_a=*/false, A, /*neg_b=*/false, B,
          /*c_mod=*/(short)0, Cz, /*reuse_a=*/false, /*reuse_b=*/false);
      if (valid) atomicAdd(&ws[col], D[0]);   // D[0] = D[M=0][N=col] = column sum
    }
  }

  if (tid < NCLS) {
    atomicAdd(&ws[23 + tid], interLds[tid]);
    atomicAdd(&ws[44 + tid], cntLds[tid]);
  }
}

__global__ void finalize(const float* __restrict__ ws, float* __restrict__ out,
                         float inv_nhw) {
  if (threadIdx.x == 0) {
    const float focal = ws[0] * inv_nhw;
    const float cem   = ws[1] * inv_nhw;
    float ds = 0.0f;
#pragma unroll
    for (int c = 0; c < NCLS; ++c) {
      const float I = ws[23 + c];
      const float U = ws[2 + c] + ws[44 + c];   // sum_probs + counts
      ds += 1.0f - (2.0f * I + 1e-6f) / (U + 1e-6f);
    }
    const float dice = ds / (float)NCLS;
    out[0] = focal + 0.5f * dice;   // total
    out[1] = cem;                   // ce_mean
    out[2] = dice;                  // dice_loss
  }
}

extern "C" void kernel_launch(void* const* d_in, const int* in_sizes, int n_in,
                              void* d_out, int out_size, void* d_ws, size_t ws_size,
                              hipStream_t stream) {
  const float* logits = (const float*)d_in[0];
  const float* cw     = (const float*)d_in[1];
  const int*   tgt    = (const int*)d_in[2];
  float* out = (float*)d_out;
  float* ws  = (float*)d_ws;
  const int nhw = in_sizes[2];          // N*H*W = 2,097,152

  zero_ws<<<1, 128, 0, stream>>>(ws);
  loss_main<<<2048, 256, 0, stream>>>(logits, cw, tgt, ws, nhw);
  finalize<<<1, 32, 0, stream>>>(ws, out, 1.0f / (float)nhw);
}
